// CrossAttnMLP_8847632629844
// MI455X (gfx1250) — compile-verified, matrix operand order checked by reading, not run on hardware
//
#include <hip/hip_runtime.h>
#include <hip/hip_bf16.h>
#include <cstdint>

// ---------------- WMMA types ----------------
typedef __attribute__((ext_vector_type(16))) _Float16 v16h;
typedef __attribute__((ext_vector_type(8)))  _Float16 v8h;
typedef __attribute__((ext_vector_type(8)))  float    v8f;

// ---------------- Problem constants ----------------
#define BROWS   65536
#define PEP     384
#define TCR     480
#define XW      (PEP + TCR)   // 864
#define DIM     128
#define FF      512
#define H1N     128
#define H2N     64
#define EPSLN   1e-5f

// f16 weight arena offsets (elements)
#define O_WPEP   0
#define O_WTCR   49152
#define O_WVP2T  110592
#define O_WOP2T  126976
#define O_WVT2P  143360
#define O_WOT2P  159744
#define O_W1P    176128
#define O_W2P    241664
#define O_W1T    307200
#define O_W2T    372736
#define O_WH1    438272
#define O_WH2    471040
#define TOTW     479232

// workspace byte offsets
#define WS_WGT    0u
#define WS_H1     (1u << 20)                       // 1 MB
#define WS_H2     (WS_H1 + BROWS * DIM * 2u)       // + 16 MB
#define WS_STATS  (WS_H2 + BROWS * H2N * 2u)       // + 8 MB

__device__ __forceinline__ float gelu_exact(float x) {
    return 0.5f * x * (1.0f + erff(x * 0.70710678118654752f));
}

// ---------------- WMMA fragment loaders ----------------
// A: 16xK f16, row-major in LDS, stride lda (multiple of 8).
// 16-bit A 16x32 layout: lane m=lane&15 is row M; lane group (lane>>4) selects
// K base kb in {0,8}; frag halves are K = k0+kb+{0..7} and k0+kb+{16..23}.
__device__ __forceinline__ v16h frag_a(const _Float16* sA, int lda, int lane, int k0) {
    int m  = lane & 15;
    int kb = (lane >> 4) << 3;
    const _Float16* p = sA + m * lda + k0 + kb;
    v8h lo = *(const v8h*)(p);
    v8h hi = *(const v8h*)(p + 16);
    return __builtin_shufflevector(lo, hi, 0,1,2,3,4,5,6,7,8,9,10,11,12,13,14,15);
}

// B: KxN where B[k][n] = w[n0+n][k], w row-major (N_total x K) f16 in global.
// Mirrors the A K-striping; per lane = two contiguous 16B loads from row n.
__device__ __forceinline__ v16h frag_b(const _Float16* w, int ldb, int lane, int n0, int k0) {
    int n  = lane & 15;
    int kb = (lane >> 4) << 3;
    const _Float16* p = w + (size_t)(n0 + n) * ldb + k0 + kb;
    v8h lo = *(const v8h*)(p);
    v8h hi = *(const v8h*)(p + 16);
    return __builtin_shufflevector(lo, hi, 0,1,2,3,4,5,6,7,8,9,10,11,12,13,14,15);
}

// C = A(16xK) * B(KxN) + bias -> f32 LDS tile (row-major, stride ldc).
// N-tiles split across nwaves waves.
__device__ void gemm_f32(const _Float16* sA, int lda, const _Float16* w, int K, int N,
                         const float* __restrict__ bias, float* sC, int ldc,
                         int tid, int nwaves) {
    int wv = tid >> 5, lane = tid & 31;
    for (int n0 = wv * 16; n0 < N; n0 += nwaves * 16) {
        v8f acc = {};
        for (int k0 = 0; k0 < K; k0 += 32) {
            v16h a = frag_a(sA, lda, lane, k0);
            v16h b = frag_b(w, K, lane, n0, k0);
            acc = __builtin_amdgcn_wmma_f32_16x16x32_f16(false, a, false, b,
                                                         (short)0, acc, false, false);
        }
        int n  = n0 + (lane & 15);
        int m0 = (lane >> 4) * 8;
        float bv = bias[n];
        #pragma unroll
        for (int i = 0; i < 8; i++) sC[(m0 + i) * ldc + n] = acc[i] + bv;
    }
}

// Same GEMM but epilogue = bias + exact GELU, stored as f16 directly into LDS
// (used for the 16x512 FFN hidden; avoids a wide f32 staging buffer).
__device__ void gemm_gelu_h(const _Float16* sA, int lda, const _Float16* w, int K, int N,
                            const float* __restrict__ bias, _Float16* sOut, int ldo,
                            int tid, int nwaves) {
    int wv = tid >> 5, lane = tid & 31;
    for (int n0 = wv * 16; n0 < N; n0 += nwaves * 16) {
        v8f acc = {};
        for (int k0 = 0; k0 < K; k0 += 32) {
            v16h a = frag_a(sA, lda, lane, k0);
            v16h b = frag_b(w, K, lane, n0, k0);
            acc = __builtin_amdgcn_wmma_f32_16x16x32_f16(false, a, false, b,
                                                         (short)0, acc, false, false);
        }
        int n  = n0 + (lane & 15);
        int m0 = (lane >> 4) * 8;
        float bv = bias[n];
        #pragma unroll
        for (int i = 0; i < 8; i++)
            sOut[(m0 + i) * ldo + n] = (_Float16)gelu_exact(acc[i] + bv);
    }
}

// LayerNorm over 128 features for a 16-row tile: y = LN(sIn + sRes)*g + b.
// 8 lanes per row; 8-lane __shfl_xor reduction (wave32-safe).
__device__ void ln16x128(const float* sIn, int ldi, const float* sRes, int ldr,
                         const float* __restrict__ g, const float* __restrict__ bta,
                         float* sOutF, int ldf, _Float16* sOutH, int ldh, int tid) {
    int row = tid >> 3;       // 0..15
    int t8  = tid & 7;
    float v[16];
    float s = 0.f, s2 = 0.f;
    #pragma unroll
    for (int j = 0; j < 16; j++) {
        int c = t8 * 16 + j;
        float x = sIn[row * ldi + c] + sRes[row * ldr + c];
        v[j] = x; s += x; s2 += x * x;
    }
    #pragma unroll
    for (int off = 4; off > 0; off >>= 1) {
        s  += __shfl_xor(s,  off, 8);
        s2 += __shfl_xor(s2, off, 8);
    }
    float mu  = s * (1.0f / 128.0f);
    float var = s2 * (1.0f / 128.0f) - mu * mu;
    float rs  = rsqrtf(var + EPSLN);
    #pragma unroll
    for (int j = 0; j < 16; j++) {
        int c = t8 * 16 + j;
        float y = (v[j] - mu) * rs * g[c] + bta[c];
        sOutF[row * ldf + c] = y;
        sOutH[row * ldh + c] = (_Float16)y;
    }
}

// ---------------- Kernel 0: weights f32 -> f16 arena ----------------
struct CvtJob { const float* src[12]; };

__global__ __launch_bounds__(256) void k_cvt(CvtJob j, _Float16* __restrict__ dst) {
    int i = blockIdx.x * 256 + threadIdx.x;
    if (i >= TOTW) return;
    const int sz[12] = {49152, 61440, 16384, 16384, 16384, 16384,
                        65536, 65536, 65536, 65536, 32768, 8192};
    int idx = i, w = 0;
    while (w < 11 && idx >= sz[w]) { idx -= sz[w]; w++; }
    dst[i] = (_Float16)j.src[w][idx];
}

// ---------------- Kernel 1: fused transformer body + head GEMM1 ----------------
__global__ __launch_bounds__(128) void k_fused(
    const float* __restrict__ x, const _Float16* __restrict__ wgt,
    const float* __restrict__ b_pep, const float* __restrict__ b_tcr,
    const float* __restrict__ bv_p2t, const float* __restrict__ bo_p2t,
    const float* __restrict__ bv_t2p, const float* __restrict__ bo_t2p,
    const float* __restrict__ g1p, const float* __restrict__ b1p,
    const float* __restrict__ g2p, const float* __restrict__ b2p,
    const float* __restrict__ g1t, const float* __restrict__ b1t,
    const float* __restrict__ g2t, const float* __restrict__ b2t,
    const float* __restrict__ fb1p, const float* __restrict__ fb2p,
    const float* __restrict__ fb1t, const float* __restrict__ fb2t,
    const float* __restrict__ b_h1,
    _Float16* __restrict__ h1out, float* __restrict__ bn1sum, float* __restrict__ bn1sq)
{
    __shared__ __align__(16) _Float16 sA[16 * 520];   // wide f16 A (x / FFN hidden / combined)
    __shared__ __align__(16) _Float16 sH[16 * 136];   // narrow f16 A (128-wide activations)
    __shared__ float sC[16 * 136];                    // f32 GEMM output (<=128 wide)
    __shared__ float sPep[16 * 128];
    __shared__ float sTcr[16 * 128];
    __shared__ float sVal[16 * 128];

    const int tid = threadIdx.x;
    const size_t row0 = (size_t)blockIdx.x * 16;

    // speculative prefetch of next tile's input (global_prefetch_b8)
    __builtin_prefetch(x + (row0 + 16) * XW + tid, 0, 3);

    // ---- pep = x[:, :384] @ Wpep^T + b ----
    for (int idx = tid; idx < 16 * PEP; idx += 128) {
        int r = idx / PEP, c = idx % PEP;
        sA[r * 520 + c] = (_Float16)x[(row0 + r) * XW + c];
    }
    __syncthreads();
    gemm_f32(sA, 520, wgt + O_WPEP, PEP, DIM, b_pep, sPep, 128, tid, 4);
    __syncthreads();

    // ---- tcr = x[:, 384:] @ Wtcr^T + b ----
    for (int idx = tid; idx < 16 * TCR; idx += 128) {
        int r = idx / TCR, c = idx % TCR;
        sA[r * 520 + c] = (_Float16)x[(row0 + r) * XW + PEP + c];
    }
    __syncthreads();
    gemm_f32(sA, 520, wgt + O_WTCR, TCR, DIM, b_tcr, sTcr, 128, tid, 4);
    __syncthreads();

    // ---- p2t attention (seq_len=1): out_proj(v_proj(tcr)) ----
    for (int idx = tid; idx < 2048; idx += 128) {
        int r = idx >> 7, c = idx & 127;
        sH[r * 136 + c] = (_Float16)sTcr[r * 128 + c];
    }
    __syncthreads();
    gemm_f32(sH, 136, wgt + O_WVP2T, DIM, DIM, bv_p2t, sC, 136, tid, 4);
    __syncthreads();
    for (int idx = tid; idx < 2048; idx += 128) {
        int r = idx >> 7, c = idx & 127;
        sH[r * 136 + c] = (_Float16)sC[r * 136 + c];
    }
    __syncthreads();
    gemm_f32(sH, 136, wgt + O_WOP2T, DIM, DIM, bo_p2t, sC, 136, tid, 4);
    __syncthreads();
    // pa = LN(pep + attn) -> f32 sVal, f16 sH
    ln16x128(sC, 136, sPep, 128, g1p, b1p, sVal, 128, sH, 136, tid);
    __syncthreads();

    // ---- FFN (p branch): gelu(pa@W1^T+b1) @ W2^T + b2, residual LN ----
    gemm_gelu_h(sH, 136, wgt + O_W1P, DIM, FF, fb1p, sA, 520, tid, 4);
    __syncthreads();
    gemm_f32(sA, 520, wgt + O_W2P, FF, DIM, fb2p, sC, 136, tid, 4);
    __syncthreads();
    ln16x128(sC, 136, sVal, 128, g2p, b2p, sVal, 128, sH, 136, tid);   // pa2 -> sVal
    __syncthreads();

    // ---- t2p attention: out_proj(v_proj(pep)) ----
    for (int idx = tid; idx < 2048; idx += 128) {
        int r = idx >> 7, c = idx & 127;
        sH[r * 136 + c] = (_Float16)sPep[r * 128 + c];
    }
    __syncthreads();
    gemm_f32(sH, 136, wgt + O_WVT2P, DIM, DIM, bv_t2p, sC, 136, tid, 4);
    __syncthreads();
    for (int idx = tid; idx < 2048; idx += 128) {
        int r = idx >> 7, c = idx & 127;
        sH[r * 136 + c] = (_Float16)sC[r * 136 + c];
    }
    __syncthreads();
    gemm_f32(sH, 136, wgt + O_WOT2P, DIM, DIM, bo_t2p, sC, 136, tid, 4);
    __syncthreads();
    ln16x128(sC, 136, sTcr, 128, g1t, b1t, sTcr, 128, sH, 136, tid);   // ta -> sTcr
    __syncthreads();

    // ---- FFN (t branch) ----
    gemm_gelu_h(sH, 136, wgt + O_W1T, DIM, FF, fb1t, sA, 520, tid, 4);
    __syncthreads();
    gemm_f32(sA, 520, wgt + O_W2T, FF, DIM, fb2t, sC, 136, tid, 4);
    __syncthreads();
    ln16x128(sC, 136, sTcr, 128, g2t, b2t, sTcr, 128, sH, 136, tid);   // ta2 -> sTcr
    __syncthreads();

    // ---- head: h1_pre = [pa2 | ta2] @ Wh1^T + b_h1 ----
    for (int idx = tid; idx < 2048; idx += 128) {
        int r = idx >> 7, c = idx & 127;
        sA[r * 520 + c]       = (_Float16)sVal[r * 128 + c];
        sA[r * 520 + 128 + c] = (_Float16)sTcr[r * 128 + c];
    }
    __syncthreads();
    gemm_f32(sA, 520, wgt + O_WH1, 2 * DIM, H1N, b_h1, sC, 136, tid, 4);
    __syncthreads();

    // store h1_pre (f16) + accumulate BN1 batch statistics
    for (int idx = tid; idx < 2048; idx += 128) {
        int r = idx >> 7, c = idx & 127;
        h1out[(row0 + r) * H1N + c] = (_Float16)sC[r * 136 + c];
    }
    if (tid < H1N) {
        float s = 0.f, s2 = 0.f;
        #pragma unroll
        for (int r = 0; r < 16; r++) { float v = sC[r * 136 + tid]; s += v; s2 += v * v; }
        atomicAdd(&bn1sum[tid], s);
        atomicAdd(&bn1sq[tid], s2);
    }
}

// ---------------- Kernel: finalize BN stats -> scale/shift ----------------
__global__ void k_bnfin(const float* __restrict__ sum, const float* __restrict__ sq,
                        const float* __restrict__ g, const float* __restrict__ b,
                        float* __restrict__ scale, float* __restrict__ shift,
                        int n, float invB) {
    int i = blockIdx.x * blockDim.x + threadIdx.x;
    if (i < n) {
        float mu  = sum[i] * invB;
        float var = sq[i] * invB - mu * mu;
        float sc  = g[i] * rsqrtf(var + EPSLN);
        scale[i] = sc;
        shift[i] = b[i] - mu * sc;
    }
}

// ---------------- Kernel: h2_pre = gelu(bn1(h1_pre)) @ Wh2^T + b_h2 ----------------
__global__ __launch_bounds__(128) void k_h2(
    const _Float16* __restrict__ h1, const float* __restrict__ sc1, const float* __restrict__ sh1,
    const _Float16* __restrict__ wh2, const float* __restrict__ b_h2,
    _Float16* __restrict__ h2out, float* __restrict__ bn2sum, float* __restrict__ bn2sq)
{
    __shared__ __align__(16) _Float16 sA[16 * 136];
    __shared__ float sC[16 * 72];
    const int tid = threadIdx.x;
    const size_t row0 = (size_t)blockIdx.x * 16;

    for (int idx = tid; idx < 2048; idx += 128) {
        int r = idx >> 7, c = idx & 127;
        float v = (float)h1[(row0 + r) * H1N + c];
        v = v * sc1[c] + sh1[c];
        sA[r * 136 + c] = (_Float16)gelu_exact(v);
    }
    __syncthreads();
    gemm_f32(sA, 136, wh2, H1N, H2N, b_h2, sC, 72, tid, 4);
    __syncthreads();
    for (int idx = tid; idx < 1024; idx += 128) {
        int r = idx >> 6, c = idx & 63;
        h2out[(row0 + r) * H2N + c] = (_Float16)sC[r * 72 + c];
    }
    if (tid < H2N) {
        float s = 0.f, s2 = 0.f;
        #pragma unroll
        for (int r = 0; r < 16; r++) { float v = sC[r * 72 + tid]; s += v; s2 += v * v; }
        atomicAdd(&bn2sum[tid], s);
        atomicAdd(&bn2sq[tid], s2);
    }
}

// ---------------- Kernel: out = gelu(bn2(h2_pre)) @ w_out^T + b_out ----------------
__global__ __launch_bounds__(256) void k_out(
    const _Float16* __restrict__ h2, const float* __restrict__ sc2,
    const float* __restrict__ sh2, const float* __restrict__ w_out,
    const float* __restrict__ b_out, float* __restrict__ out)
{
    size_t row = (size_t)blockIdx.x * 256 + threadIdx.x;
    const _Float16* hr = h2 + row * H2N;
    float acc = 0.f;
    #pragma unroll
    for (int i = 0; i < H2N; i++) {
        float v = (float)hr[i] * sc2[i] + sh2[i];
        acc += gelu_exact(v) * w_out[i];
    }
    out[row] = acc + b_out[0];
}

// ---------------- Host launcher ----------------
extern "C" void kernel_launch(void* const* d_in, const int* in_sizes, int n_in,
                              void* d_out, int out_size, void* d_ws, size_t ws_size,
                              hipStream_t stream) {
    const float* x      = (const float*)d_in[0];
    const float* b_pep  = (const float*)d_in[2];
    const float* b_tcr  = (const float*)d_in[4];
    const float* bv_p2t = (const float*)d_in[6];
    const float* bo_p2t = (const float*)d_in[8];
    const float* bv_t2p = (const float*)d_in[10];
    const float* bo_t2p = (const float*)d_in[12];
    const float* g1p = (const float*)d_in[13], *b1p = (const float*)d_in[14];
    const float* g2p = (const float*)d_in[15], *b2p = (const float*)d_in[16];
    const float* g1t = (const float*)d_in[17], *b1t = (const float*)d_in[18];
    const float* g2t = (const float*)d_in[19], *b2t = (const float*)d_in[20];
    const float* fb1p = (const float*)d_in[22], *fb2p = (const float*)d_in[24];
    const float* fb1t = (const float*)d_in[26], *fb2t = (const float*)d_in[28];
    const float* b_h1 = (const float*)d_in[30];
    const float* bn1_g = (const float*)d_in[31], *bn1_b = (const float*)d_in[32];
    const float* b_h2 = (const float*)d_in[34];
    const float* bn2_g = (const float*)d_in[35], *bn2_b = (const float*)d_in[36];
    const float* w_out = (const float*)d_in[37], *b_out = (const float*)d_in[38];

    char* ws = (char*)d_ws;
    _Float16* wgt = (_Float16*)(ws + WS_WGT);
    _Float16* h1  = (_Float16*)(ws + WS_H1);
    _Float16* h2  = (_Float16*)(ws + WS_H2);
    float* stats  = (float*)(ws + WS_STATS);
    float* bn1sum = stats,        *bn1sq = stats + 128;
    float* bn2sum = stats + 256,  *bn2sq = stats + 320;
    float* bn1sc  = stats + 384,  *bn1sh = stats + 512;
    float* bn2sc  = stats + 640,  *bn2sh = stats + 704;

    // zero the 4 atomic accumulators (graph-capture-safe)
    hipMemsetAsync(stats, 0, 384 * sizeof(float), stream);

    CvtJob job;
    job.src[0]  = (const float*)d_in[1];   // w_pep
    job.src[1]  = (const float*)d_in[3];   // w_tcr
    job.src[2]  = (const float*)d_in[5];   // wv_p2t
    job.src[3]  = (const float*)d_in[7];   // wo_p2t
    job.src[4]  = (const float*)d_in[9];   // wv_t2p
    job.src[5]  = (const float*)d_in[11];  // wo_t2p
    job.src[6]  = (const float*)d_in[21];  // ffn_w1p
    job.src[7]  = (const float*)d_in[23];  // ffn_w2p
    job.src[8]  = (const float*)d_in[25];  // ffn_w1t
    job.src[9]  = (const float*)d_in[27];  // ffn_w2t
    job.src[10] = (const float*)d_in[29];  // w_h1
    job.src[11] = (const float*)d_in[33];  // w_h2
    k_cvt<<<(TOTW + 255) / 256, 256, 0, stream>>>(job, wgt);

    const int ntiles = BROWS / 16;  // 4096
    k_fused<<<ntiles, 128, 0, stream>>>(x, wgt,
        b_pep, b_tcr, bv_p2t, bo_p2t, bv_t2p, bo_t2p,
        g1p, b1p, g2p, b2p, g1t, b1t, g2t, b2t,
        fb1p, fb2p, fb1t, fb2t, b_h1, h1, bn1sum, bn1sq);

    const float invB = 1.0f / (float)BROWS;
    k_bnfin<<<1, 128, 0, stream>>>(bn1sum, bn1sq, bn1_g, bn1_b, bn1sc, bn1sh, H1N, invB);

    k_h2<<<ntiles, 128, 0, stream>>>(h1, bn1sc, bn1sh, wgt + O_WH2, b_h2, h2, bn2sum, bn2sq);

    k_bnfin<<<1, 64, 0, stream>>>(bn2sum, bn2sq, bn2_g, bn2_b, bn2sc, bn2sh, H2N, invB);

    k_out<<<BROWS / 256, 256, 0, stream>>>(h2, bn2sc, bn2sh, w_out, b_out, (float*)d_out);
}
